// KL_div_loss_with_knn_25761213841509
// MI455X (gfx1250) — matrix operand order repaired and checked by reading, not compiled, not run on hardware
//
#include <hip/hip_runtime.h>
#include <math.h>

typedef __attribute__((ext_vector_type(16))) __bf16 v16bf;
typedef __attribute__((ext_vector_type(8)))  __bf16 v8bf;
typedef __attribute__((ext_vector_type(8)))  float  v8f;
typedef __attribute__((ext_vector_type(4)))  int    v4i;

#define DFT     256   // feature dimension (fixed by reference)
#define BSTRIDE 264   // 256 + 8 pad elements (16B) -> conflict-free ds_load_b128
#define NTILE   64    // shared column tile per iteration
#define MROWS   256   // rows per block: 8 waves x 32 (2 row-tiles per wave)
#define COLSPLIT 16   // column splits per pass (grid parallelism)

// ---------- async global -> LDS copy (CDNA5: GLOBAL_LOAD_ASYNC_TO_LDS_B128, ASYNCcnt) ----------
#if __has_builtin(__builtin_amdgcn_global_load_async_to_lds_b128)
#define HAVE_ASYNC_LDS 1
#define AS1 __attribute__((address_space(1)))
#define AS3 __attribute__((address_space(3)))
#define ASYNC_CP_B128(lptr, gptr)                                               \
    __builtin_amdgcn_global_load_async_to_lds_b128((AS1 v4i*)(void*)(gptr),     \
                                                   (AS3 v4i*)(void*)(lptr), 0, 0)
#else
#define HAVE_ASYNC_LDS 0
#endif

__device__ __forceinline__ void wait_async0() {
#if __has_builtin(__builtin_amdgcn_s_wait_asynccnt)
    __builtin_amdgcn_s_wait_asynccnt(0);
#elif HAVE_ASYNC_LDS
    asm volatile("s_wait_asynccnt 0x0" ::: "memory");
#endif
}

__device__ __forceinline__ void load_frag(v16bf& f, const __bf16* base, int ofs) {
    ((v8bf*)&f)[0] = *(const v8bf*)(base + ofs);
    ((v8bf*)&f)[1] = *(const v8bf*)(base + ofs + 16);
}

// ---- 1) precompute: bf16 hi/lo split planes + fp32 row sq-norms + init min arrays ----
__global__ void __launch_bounds__(256)
prep_kernel(const float* __restrict__ s1, const float* __restrict__ s2,
            __bf16* s1h, __bf16* s1l, __bf16* s2h, __bf16* s2l,
            float* sq1, float* sq2, float* numin, float* rhomin, int N, int M) {
    const int tid  = threadIdx.x;
    const int lane = tid & 31;
    const int wave = tid >> 5;
    const int gw   = blockIdx.x * 8 + wave;       // one wave per row
    const int gid  = blockIdx.x * 256 + tid;
    if (gid < N) { numin[gid] = __builtin_inff(); rhomin[gid] = __builtin_inff(); }

    const float* src; __bf16 *dh, *dl; float* sq; int row;
    if (gw < N)          { src = s1; dh = s1h; dl = s1l; sq = sq1; row = gw; }
    else if (gw < N + M) { src = s2; dh = s2h; dl = s2l; sq = sq2; row = gw - N; }
    else return;

    const float* rp = src + (size_t)row * DFT;
    float ssum = 0.f;
#pragma unroll
    for (int e = 0; e < DFT / 32; ++e) {
        const int c = e * 32 + lane;
        const float x = rp[c];
        ssum += x * x;
        const __bf16 h = (__bf16)x;
        const __bf16 l = (__bf16)(x - (float)h);
        dh[(size_t)row * DFT + c] = h;
        dl[(size_t)row * DFT + c] = l;
    }
#pragma unroll
    for (int m = 16; m >= 1; m >>= 1) ssum += __shfl_xor(ssum, m, 32);
    if (lane == 0) sq[row] = ssum;
}

// ---- 2) fused distance-GEMM + row-min.  pass 0: nu (vs s2), pass 1: rho (vs s1, skip diag) ----
// Workgroup tile: 256 rows (8 waves x 2 row-tiles) x 64 shared cols (double-buffered async LDS B).
__global__ void __launch_bounds__(256)
knn_min_kernel(const __bf16* __restrict__ s1h, const __bf16* __restrict__ s1l,
               const __bf16* __restrict__ s2h, const __bf16* __restrict__ s2l,
               const float* __restrict__ sq1, const float* __restrict__ sq2,
               float* numin, float* rhomin, int N, int M, int blocks_per_pass) {
    __shared__ __attribute__((aligned(16))) __bf16 BsH[2][NTILE * BSTRIDE];
    __shared__ __attribute__((aligned(16))) __bf16 BsL[2][NTILE * BSTRIDE];

    const int bid  = (int)blockIdx.x;
    const int pass = (bid >= blocks_per_pass) ? 1 : 0;
    const int pbid = pass ? bid - blocks_per_pass : bid;
    const int row_blocks = N / MROWS;
    const int row_block  = pbid % row_blocks;
    const int col_split  = pbid / row_blocks;

    const __bf16* Bh  = pass ? s1h : s2h;
    const __bf16* Bl  = pass ? s1l : s2l;
    const float*  sqy = pass ? sq1 : sq2;
    float*        omin = pass ? rhomin : numin;
    const int ncols = pass ? N : M;
    const int cols_per_split = ncols / COLSPLIT;
    const int col0  = col_split * cols_per_split;
    const int iters = cols_per_split / NTILE;

    const int tid  = threadIdx.x;
    const int lane = tid & 31;
    const int nsub = lane & 15;
    const int sel  = lane >> 4;
    const int rsel = sel << 3;                       // C layout: lanes 16-31 hold rows M=8..15
    const int wrow_base = row_block * MROWS + (tid >> 5) * 32;   // this wave's 32 rows

    // staging map: thread -> (col 0..63, 64-element chunk 0..3), bijective over 256 threads
    const int scol   = tid & 63;
    const int schunk = (tid >> 6) * 64;

    // stage one 64-col tile (hi+lo) into LDS buffer `buf` (this thread's 256B share)
    auto stage = [&](int buf, int colbase) {
        const __bf16* gh = Bh + (size_t)(colbase + scol) * DFT + schunk;
        const __bf16* gl = Bl + (size_t)(colbase + scol) * DFT + schunk;
        __bf16* lh = &BsH[buf][scol * BSTRIDE + schunk];
        __bf16* ll = &BsL[buf][scol * BSTRIDE + schunk];
#if HAVE_ASYNC_LDS
#pragma unroll
        for (int j = 0; j < 8; ++j) ASYNC_CP_B128(lh + j * 8, gh + j * 8);
#pragma unroll
        for (int j = 0; j < 8; ++j) ASYNC_CP_B128(ll + j * 8, gl + j * 8);
#else
        v8bf th[8], tl[8];
#pragma unroll
        for (int j = 0; j < 8; ++j) th[j] = *(const v8bf*)(gh + j * 8);
#pragma unroll
        for (int j = 0; j < 8; ++j) tl[j] = *(const v8bf*)(gl + j * 8);
#pragma unroll
        for (int j = 0; j < 8; ++j) *(v8bf*)(lh + j * 8) = th[j];
#pragma unroll
        for (int j = 0; j < 8; ++j) *(v8bf*)(ll + j * 8) = tl[j];
#endif
    };

    // ---- resident A fragments: 2 row-tiles x 16 rows, K=0..255, hi+lo (256 VGPRs) ----
    v16bf a_h[16], a_l[16];     // [m2*8 + kc]
#pragma unroll
    for (int m2 = 0; m2 < 2; ++m2) {
        const __bf16* pAh = s1h + (size_t)(wrow_base + m2 * 16 + nsub) * DFT + sel * 8;
        const __bf16* pAl = s1l + (size_t)(wrow_base + m2 * 16 + nsub) * DFT + sel * 8;
#pragma unroll
        for (int kc = 0; kc < 8; ++kc) {
            ((v8bf*)&a_h[m2 * 8 + kc])[0] = *(const v8bf*)(pAh + kc * 32);
            ((v8bf*)&a_h[m2 * 8 + kc])[1] = *(const v8bf*)(pAh + kc * 32 + 16);
            ((v8bf*)&a_l[m2 * 8 + kc])[0] = *(const v8bf*)(pAl + kc * 32);
            ((v8bf*)&a_l[m2 * 8 + kc])[1] = *(const v8bf*)(pAl + kc * 32 + 16);
        }
    }

    float rm[2][8];
#pragma unroll
    for (int m2 = 0; m2 < 2; ++m2)
#pragma unroll
        for (int r = 0; r < 8; ++r) rm[m2][r] = __builtin_inff();

    // LDS fragment base: col = t*16 + nsub, elem offset kc*32 + sel*8 (+16 for 2nd half) (B 32x16 layout)
    const int lbase = nsub * BSTRIDE + sel * 8;

    // prologue: fill buffer 0
    stage(0, col0);
    wait_async0();
    __syncthreads();

    int cur = 0;
    for (int it = 0; it < iters; ++it) {
        const int colbase = col0 + it * NTILE;
        if (it + 1 < iters) stage(cur ^ 1, colbase + NTILE);   // overlap with compute

        const __bf16* bH = &BsH[cur][0];
        const __bf16* bL = &BsL[cur][0];

        // ---- compute: 192 WMMAs (2 row-tiles x split-bf16), B frags pipelined 1 group ahead ----
        v8f acc[2][4] = {};
        v16bf bh[2], bl[2];
        load_frag(bh[0], bH, lbase);
        load_frag(bl[0], bL, lbase);
#pragma unroll
        for (int g = 0; g < 32; ++g) {          // g = kc*4 + t
            const int kc = g >> 2, t = g & 3;
            const int cb = g & 1, nb = cb ^ 1;
            if (g + 1 < 32) {
                const int g1 = g + 1;
                const int ofs1 = lbase + (g1 & 3) * 16 * BSTRIDE + (g1 >> 2) * 32;
                load_frag(bh[nb], bH, ofs1);
                load_frag(bl[nb], bL, ofs1);
            }
#pragma unroll
            for (int m2 = 0; m2 < 2; ++m2) {
                acc[m2][t] = __builtin_amdgcn_wmma_f32_16x16x32_bf16(false, a_h[m2 * 8 + kc], false, bh[cb], (short)0, acc[m2][t], false, false);
                acc[m2][t] = __builtin_amdgcn_wmma_f32_16x16x32_bf16(false, a_h[m2 * 8 + kc], false, bl[cb], (short)0, acc[m2][t], false, false);
                acc[m2][t] = __builtin_amdgcn_wmma_f32_16x16x32_bf16(false, a_l[m2 * 8 + kc], false, bh[cb], (short)0, acc[m2][t], false, false);
            }
        }

        // ---- fold min_j ( ||y_j||^2 - 2 x_i.y_j )  (VALU, overlaps async staging) ----
#pragma unroll
        for (int t = 0; t < 4; ++t) {
            const int col = colbase + t * 16 + nsub;
            const float sy = sqy[col];
#pragma unroll
            for (int m2 = 0; m2 < 2; ++m2)
#pragma unroll
                for (int r = 0; r < 8; ++r) {
                    float v = fmaf(-2.f, acc[m2][t][r], sy);
                    if (pass && col == (wrow_base + m2 * 16 + rsel + r)) v = __builtin_inff();
                    rm[m2][r] = fminf(rm[m2][r], v);
                }
        }

        wait_async0();       // my async copies for the next tile have landed
        __syncthreads();     // everyone's copies landed; everyone done reading `cur`
        cur ^= 1;
    }

    // reduce over the 16 lanes of each half (columns); halves hold different rows
#pragma unroll
    for (int m = 8; m >= 1; m >>= 1)
#pragma unroll
        for (int m2 = 0; m2 < 2; ++m2)
#pragma unroll
            for (int r = 0; r < 8; ++r)
                rm[m2][r] = fminf(rm[m2][r], __shfl_xor(rm[m2][r], m, 32));

    if (nsub == 0) {                 // lane 0 -> rows 0..7, lane 16 -> rows 8..15 (per row-tile)
#pragma unroll
        for (int m2 = 0; m2 < 2; ++m2)
#pragma unroll
            for (int r = 0; r < 8; ++r) {
                const int row = wrow_base + m2 * 16 + rsel + r;
                float v = fmaxf(rm[m2][r] + sq1[row], 0.f);   // d^2 = sqx + (sqy - 2dot), clamp
                atomicMin((unsigned int*)(omin + row), __float_as_uint(v));  // valid: v >= 0
            }
    }
}

// ---- 3) final scalar:  log(m/(n-1)) + (d/n) * sum_i log(nu_i / rho_i) ----
__global__ void finalize_kernel(const float* __restrict__ numin, const float* __restrict__ rhomin,
                                float* out, int N, int M) {
    __shared__ float sbuf[256];
    const int tid = threadIdx.x;
    float s = 0.f;
    for (int i = tid; i < N; i += 256) {
        const float nu  = sqrtf(numin[i]);
        float rho = sqrtf(rhomin[i]);
        if (rho == 0.f) rho = 1e-10f;
        s += logf(nu / rho);
    }
    sbuf[tid] = s; __syncthreads();
    for (int k = 128; k > 0; k >>= 1) { if (tid < k) sbuf[tid] += sbuf[tid + k]; __syncthreads(); }
    if (tid == 0) out[0] = logf((float)M / (float)(N - 1)) + ((float)DFT / (float)N) * sbuf[0];
}

extern "C" void kernel_launch(void* const* d_in, const int* in_sizes, int n_in,
                              void* d_out, int out_size, void* d_ws, size_t ws_size,
                              hipStream_t stream) {
    const float* s1 = (const float*)d_in[0];
    const float* s2 = (const float*)d_in[1];
    const int N = in_sizes[0] / DFT;   // 8192
    const int M = in_sizes[1] / DFT;   // 8192

    char* ws = (char*)d_ws;
    size_t o = 0;
    __bf16* s1h = (__bf16*)(ws + o); o += (size_t)N * DFT * 2;
    __bf16* s1l = (__bf16*)(ws + o); o += (size_t)N * DFT * 2;
    __bf16* s2h = (__bf16*)(ws + o); o += (size_t)M * DFT * 2;
    __bf16* s2l = (__bf16*)(ws + o); o += (size_t)M * DFT * 2;
    float*  sq1 = (float*)(ws + o);  o += (size_t)N * 4;
    float*  sq2 = (float*)(ws + o);  o += (size_t)M * 4;
    float*  numin  = (float*)(ws + o); o += (size_t)N * 4;
    float*  rhomin = (float*)(ws + o); o += (size_t)N * 4;

    const int prep_blocks = (N + M) / 8;           // one wave per row, 8 waves/block
    prep_kernel<<<prep_blocks, 256, 0, stream>>>(s1, s2, s1h, s1l, s2h, s2l,
                                                 sq1, sq2, numin, rhomin, N, M);

    const int bpp = (N / MROWS) * COLSPLIT;        // 32 row-blocks x 16 col-splits = 512 per pass
    knn_min_kernel<<<2 * bpp, 256, 0, stream>>>(s1h, s1l, s2h, s2l, sq1, sq2,
                                                numin, rhomin, N, M, bpp);

    finalize_kernel<<<1, 256, 0, stream>>>(numin, rhomin, (float*)d_out, N, M);
}
